// ABSEncoder_84353157693655
// MI455X (gfx1250) — matrix operand order repaired and checked by reading, not compiled, not run on hardware
//
#include <hip/hip_runtime.h>

// ---------------------------------------------------------------------------
// ABS encoder attention for MI455X (gfx1250, wave32, WMMA bf16 16x16x32).
// out = softmax(mask(xe @ (yce @ Pw + Pb)^T))^T @ xe
//
// GEMMs use 64x64 register-blocked tiles per wave: 4 A-frags x 4 B-frags ->
// 16 v_wmma_f32_16x16x32_bf16 per K-step, fp32 accumulation (~21 FLOP/B from
// L2; panel re-reads cut 4x in each dimension -> ~0.6 GB L2 traffic, below
// the ~0.5 GB @ 23.3 TB/s HBM floor). __launch_bounds__(128, 1) lifts the
// default 128-VGPR occupancy cap so the 16 fp32 accumulators (128 VGPRs) and
// 8 transient bf16 fragments (64 VGPRs) live in registers with no scratch
// spills; ~224 VGPRs still allows ~4 waves/SIMD on a 1024-VGPR file.
// ---------------------------------------------------------------------------

typedef __attribute__((ext_vector_type(16))) __bf16 v16bf;
typedef __attribute__((ext_vector_type(8)))  __bf16 v8bf;
typedef __attribute__((ext_vector_type(8)))  float  v8f;

namespace {
constexpr int kB   = 32;
constexpr int kSrc = 1024;
constexpr int kSeq = 128;
constexpr int kCtx = 5;
constexpr int kD   = 512;
constexpr int kKy  = kCtx * kD;   // 2560
}

// ---- WMMA fragment helpers (contiguous K = 16*(lane/16) + s, same for A and B, wave32) ----------

// A: 16x32 bf16, row-major source with leading dim `lda` (K contiguous).
// lanes 0-15 : row = lane,    K = 0..15
// lanes 16-31: row = lane-16, K = 16..31
__device__ __forceinline__ v16bf load_a_frag(const __bf16* __restrict__ A,
                                             int lda, int mbase, int k0, int lane) {
  const int lrow  = lane & 15;
  const int lhalf = lane >> 4;
  const __bf16* p = A + (size_t)(mbase + lrow) * lda + k0 + lhalf * 16;
  return *(const v16bf*)(p);
}

// B: 32x16 bf16 taken from an N-major, K-contiguous array Bt (ldb = K stride).
// VGPR v holds K={2v,2v+1}; lanes 0-15 cover K 0..15 at col=lane,
// lanes 16-31 cover K 16..31 at col=lane-16.  => one 32B contiguous load.
__device__ __forceinline__ v16bf load_b_frag(const __bf16* __restrict__ Bt,
                                             int ldb, int nbase, int k0, int lane) {
  const int lrow  = lane & 15;
  const int lhalf = lane >> 4;
  const __bf16* p = Bt + (size_t)(nbase + lrow) * ldb + k0 + lhalf * 16;
  return *(const v16bf*)(p);
}

__device__ __forceinline__ v8f wmma_bf16(v16bf a, v16bf b, v8f c) {
  // (neg_a, A, neg_b, B, c_mod, C, reuse_a, reuse_b)
  return __builtin_amdgcn_wmma_f32_16x16x32_bf16(false, a, false, b,
                                                 (short)0, c, false, false);
}

// 64x64 output tile per wave: 4x4 grid of 16x16 WMMA tiles.
__device__ __forceinline__ void mma_tile_64x64(const __bf16* __restrict__ A, int lda,
                                               const __bf16* __restrict__ Bt, int ldb,
                                               int mbase, int nbase, int K, int lane,
                                               v8f (&c)[4][4]) {
  for (int k0 = 0; k0 < K; k0 += 32) {
    v16bf a[4], b[4];
#pragma unroll
    for (int mi = 0; mi < 4; ++mi)
      a[mi] = load_a_frag(A, lda, mbase + 16 * mi, k0, lane);
#pragma unroll
    for (int ni = 0; ni < 4; ++ni)
      b[ni] = load_b_frag(Bt, ldb, nbase + 16 * ni, k0, lane);
#pragma unroll
    for (int mi = 0; mi < 4; ++mi)
#pragma unroll
      for (int ni = 0; ni < 4; ++ni)
        c[mi][ni] = wmma_bf16(a[mi], b[ni], c[mi][ni]);
  }
}

// ---- Prep kernels ----------------------------------------------------------

// P_w [Ky, D] fp32 -> PwT [D, Ky] bf16 (K-contiguous B operand for GEMM1)
__global__ void prep_pw_kernel(const float* __restrict__ Pw,
                               __bf16* __restrict__ PwT) {
  int i = blockIdx.x * blockDim.x + threadIdx.x;
  if (i >= kKy * kD) return;
  int k = i / kD;
  int d = i % kD;
  PwT[(size_t)d * kKy + k] = (__bf16)Pw[i];
}

// yce[b][s][c*D+d] = G_emb[yc[b][s*C+c]][d]   (flat orders line up exactly)
__global__ void prep_yce_kernel(const int* __restrict__ yc,
                                const float* __restrict__ Gemb,
                                __bf16* __restrict__ yce) {
  long i = (long)blockIdx.x * blockDim.x + threadIdx.x;
  if (i >= (long)kB * kSeq * kKy) return;
  int  d = (int)(i % kD);
  long r = i / kD;                       // == flat index into yc
  int  tok = yc[r];
  yce[i] = (__bf16)Gemb[(size_t)tok * kD + d];
}

// xe[b][x][d] = F_emb[x_tok][d]; also xeT[b][d][x] (B operand for GEMM3)
__global__ void prep_xe_kernel(const int* __restrict__ x,
                               const float* __restrict__ Femb,
                               __bf16* __restrict__ xe,
                               __bf16* __restrict__ xeT) {
  long i = (long)blockIdx.x * blockDim.x + threadIdx.x;
  if (i >= (long)kB * kSrc * kD) return;
  int  d  = (int)(i % kD);
  long r  = i / kD;                      // b*kSrc + xx
  int  b  = (int)(r / kSrc);
  int  xx = (int)(r % kSrc);
  float v = Femb[(size_t)x[r] * kD + d];
  xe[i] = (__bf16)v;
  xeT[((size_t)b * kD + d) * kSrc + xx] = (__bf16)v;
}

// ---- GEMM1: py[b][s][d] = yce[b][s][:] . PwT[d][:] + Pb[d]  (M=4096,K=2560,N=512)
__global__ void __launch_bounds__(128, 1)
gemm_py_kernel(const __bf16* __restrict__ yce,
               const __bf16* __restrict__ PwT,
               const float* __restrict__ Pb,
               __bf16* __restrict__ py) {
  const int wave = blockIdx.x * (blockDim.x >> 5) + (threadIdx.x >> 5);
  const int lane = threadIdx.x & 31;
  const int tilesN = kD / 64;            // 8
  const int tm = wave / tilesN;          // over (B*SEQ)/64 = 64
  const int tn = wave % tilesN;
  const int mbase = tm * 64, nbase = tn * 64;

  v8f c[4][4] = {};
  mma_tile_64x64(yce, kKy, PwT, kKy, mbase, nbase, kKy, lane, c);

  const int lrow = lane & 15, lhalf = lane >> 4;
#pragma unroll
  for (int ni = 0; ni < 4; ++ni) {
    const int n = nbase + 16 * ni + lrow;
    const float bias = Pb[n];
#pragma unroll
    for (int mi = 0; mi < 4; ++mi)
#pragma unroll
      for (int r = 0; r < 8; ++r) {
        int m = mbase + 16 * mi + r + lhalf * 8;
        py[(size_t)m * kD + n] = (__bf16)(c[mi][ni][r] + bias);
      }
  }
}

// ---- GEMM2: logits[b][s][x] = xe[b][x][:] . py[b][s][:]  (+ mask)
// py row-major already IS the N-major/K-contiguous B operand.
__global__ void __launch_bounds__(128, 1)
attn_logits_kernel(const __bf16* __restrict__ xe,
                   const __bf16* __restrict__ py,
                   const int* __restrict__ x,
                   float* __restrict__ a_out) {
  const int wave = blockIdx.x * (blockDim.x >> 5) + (threadIdx.x >> 5);
  const int lane = threadIdx.x & 31;
  const int tilesPerBatch = (kSrc / 64) * (kSeq / 64);   // 16*2 = 32
  const int b  = wave / tilesPerBatch;
  const int t  = wave % tilesPerBatch;
  const int tm = t / (kSeq / 64);        // over SRC tiles (16)
  const int tn = t % (kSeq / 64);        // over SEQ tiles (2)
  const int mbase = tm * 64, nbase = tn * 64;

  const __bf16* A  = xe + (size_t)b * kSrc * kD;
  const __bf16* Bt = py + (size_t)b * kSeq * kD;

  v8f c[4][4] = {};
  mma_tile_64x64(A, kD, Bt, kD, mbase, nbase, kD, lane, c);

  const int lrow = lane & 15, lhalf = lane >> 4;
#pragma unroll
  for (int mi = 0; mi < 4; ++mi)
#pragma unroll
    for (int r = 0; r < 8; ++r) {
      const int xg = mbase + 16 * mi + r + lhalf * 8;
      const float maskv = (x[(size_t)b * kSrc + xg] == 0) ? -1e9f : 0.0f;
#pragma unroll
      for (int ni = 0; ni < 4; ++ni) {
        const int sg = nbase + 16 * ni + lrow;
        a_out[((size_t)b * kSeq + sg) * kSrc + xg] = c[mi][ni][r] + maskv;
      }
    }
}

// ---- Softmax over SRC per (b,s) row; emit fp32 (output) + bf16 (GEMM3 A) ---
__global__ void softmax_kernel(float* __restrict__ a_out,
                               __bf16* __restrict__ a_bf) {
  const int row = blockIdx.x;            // 0 .. B*SEQ-1
  const int t   = threadIdx.x;           // 256 threads, 4 elems each
  float* p = a_out + (size_t)row * kSrc;
  __shared__ float red[256];

  float v0 = p[t], v1 = p[t + 256], v2 = p[t + 512], v3 = p[t + 768];
  float m = fmaxf(fmaxf(v0, v1), fmaxf(v2, v3));
  red[t] = m; __syncthreads();
  for (int s = 128; s > 0; s >>= 1) {
    if (t < s) red[t] = fmaxf(red[t], red[t + s]);
    __syncthreads();
  }
  const float rowmax = red[0];
  __syncthreads();

  float e0 = __expf(v0 - rowmax), e1 = __expf(v1 - rowmax);
  float e2 = __expf(v2 - rowmax), e3 = __expf(v3 - rowmax);
  red[t] = e0 + e1 + e2 + e3; __syncthreads();
  for (int s = 128; s > 0; s >>= 1) {
    if (t < s) red[t] += red[t + s];
    __syncthreads();
  }
  const float inv = 1.0f / red[0];

  e0 *= inv; e1 *= inv; e2 *= inv; e3 *= inv;
  p[t] = e0; p[t + 256] = e1; p[t + 512] = e2; p[t + 768] = e3;
  __bf16* q = a_bf + (size_t)row * kSrc;
  q[t] = (__bf16)e0; q[t + 256] = (__bf16)e1;
  q[t + 512] = (__bf16)e2; q[t + 768] = (__bf16)e3;
}

// ---- GEMM3: out[b][s][d] = a[b][s][:] . xeT[b][d][:]  (M=128,K=1024,N=512)
__global__ void __launch_bounds__(128, 1)
attn_out_kernel(const __bf16* __restrict__ a_bf,
                const __bf16* __restrict__ xeT,
                float* __restrict__ out) {
  const int wave = blockIdx.x * (blockDim.x >> 5) + (threadIdx.x >> 5);
  const int lane = threadIdx.x & 31;
  const int tilesPerBatch = (kSeq / 64) * (kD / 64);     // 2*8 = 16
  const int b  = wave / tilesPerBatch;
  const int t  = wave % tilesPerBatch;
  const int tm = t / (kD / 64);          // over SEQ tiles (2)
  const int tn = t % (kD / 64);          // over D tiles (8)
  const int mbase = tm * 64, nbase = tn * 64;

  const __bf16* A  = a_bf + (size_t)b * kSeq * kSrc;
  const __bf16* Bt = xeT + (size_t)b * kD * kSrc;

  v8f c[4][4] = {};
  mma_tile_64x64(A, kSrc, Bt, kSrc, mbase, nbase, kSrc, lane, c);

  const int lrow = lane & 15, lhalf = lane >> 4;
#pragma unroll
  for (int mi = 0; mi < 4; ++mi)
#pragma unroll
    for (int r = 0; r < 8; ++r) {
      const int sg = mbase + 16 * mi + r + lhalf * 8;
#pragma unroll
      for (int ni = 0; ni < 4; ++ni) {
        const int dg = nbase + 16 * ni + lrow;
        out[((size_t)b * kSeq + sg) * kD + dg] = c[mi][ni][r];
      }
    }
}

// ---------------------------------------------------------------------------

extern "C" void kernel_launch(void* const* d_in, const int* in_sizes, int n_in,
                              void* d_out, int out_size, void* d_ws, size_t ws_size,
                              hipStream_t stream) {
  (void)in_sizes; (void)n_in; (void)out_size; (void)ws_size;

  const int*   x    = (const int*)d_in[0];     // [B, SRC] int32
  const int*   yc   = (const int*)d_in[1];     // [B, SEQ*CTX] int32
  const float* Femb = (const float*)d_in[2];   // [VOCAB, D]
  const float* Gemb = (const float*)d_in[3];   // [VOCAB, D]
  const float* Pw   = (const float*)d_in[4];   // [Ky, D]
  const float* Pb   = (const float*)d_in[5];   // [D]

  float* out   = (float*)d_out;                         // [B,SEQ,D]
  float* a_out = out + (size_t)kB * kSeq * kD;          // [B,SEQ,SRC]

  char* ws = (char*)d_ws;
  size_t off = 0;
  __bf16* yce = (__bf16*)(ws + off); off += (size_t)kB * kSeq * kKy * 2;  // 20 MB
  __bf16* PwT = (__bf16*)(ws + off); off += (size_t)kD * kKy * 2;         // 2.5 MB
  __bf16* xe  = (__bf16*)(ws + off); off += (size_t)kB * kSrc * kD * 2;   // 32 MB
  __bf16* xeT = (__bf16*)(ws + off); off += (size_t)kB * kSrc * kD * 2;   // 32 MB
  __bf16* py  = (__bf16*)(ws + off); off += (size_t)kB * kSeq * kD * 2;   // 4 MB
  __bf16* abf = (__bf16*)(ws + off); off += (size_t)kB * kSeq * kSrc * 2; // 8 MB

  // --- Stage 0: gathers + layout conversions (fp32 -> bf16) ---
  prep_pw_kernel<<<(kKy * kD + 255) / 256, 256, 0, stream>>>(Pw, PwT);
  prep_yce_kernel<<<(int)(((long)kB * kSeq * kKy + 255) / 256), 256, 0, stream>>>(yc, Gemb, yce);
  prep_xe_kernel<<<(int)(((long)kB * kSrc * kD + 255) / 256), 256, 0, stream>>>(x, Femb, xe, xeT);

  // --- Stage 1: py = yce @ Pw + Pb  (512 waves of 64x64 tiles) ---
  gemm_py_kernel<<<512 / 4, 128, 0, stream>>>(yce, PwT, Pb, py);

  // --- Stage 2: logits (masked), stored transposed [b][s][x] (1024 waves) ---
  attn_logits_kernel<<<1024 / 4, 128, 0, stream>>>(xe, py, x, a_out);

  // --- Stage 3: softmax over SRC per row ---
  softmax_kernel<<<kB * kSeq, 256, 0, stream>>>(a_out, abf);

  // --- Stage 4: out = a @ xe  (512 waves of 64x64 tiles) ---
  attn_out_kernel<<<512 / 4, 128, 0, stream>>>(abf, xeT, out);
}
